// Graph_Transformer_Block_71554155151907
// MI455X (gfx1250) — matrix operand order repaired and checked by reading, not compiled, not run on hardware
//
#include <hip/hip_runtime.h>
#include <hip/hip_bf16.h>

// ---------------------------------------------------------------------------
// Graph transformer block for MI455X (gfx1250), wave32 + WMMA f16->f32.
// All heavy GEMMs run on v_wmma_f32_16x16x32_f16. Weights are pre-packed to
// f16 in the per-lane B-fragment layout so each B fragment is one contiguous
// 32B global load; A tiles are staged in LDS (f16) and loaded as 2x b128.
// B fragments of a K-step are batched into independent registers before the
// WMMA burst so the loads pipeline instead of serializing on s_wait_loadcnt.
// ---------------------------------------------------------------------------

typedef __attribute__((ext_vector_type(16))) _Float16 v16h;
typedef __attribute__((ext_vector_type(8)))  _Float16 v8h;
typedef __attribute__((ext_vector_type(8)))  float    v8f;

#define H     4
#define DK    64
#define DN    128
#define DE    64
#define DNO   256
#define DOE   128
#define FIN   320          // 2*DN + DE
#define FIN2  576          // 2*DNO + DE
#define EPS   1e-6f
#define RSQRT_FIN 0.05590169943749474f   // 1/sqrt(320)

// LDS A-tile strides (halves). Must be multiples of 8 so every b128 chunk is
// 16B aligned; odd multiples of 8 also spread banks nicely.
#define SK_F   328   // >= 320
#define SK_X   264   // >= 256
#define SK_EF  584   // >= 576

// ---- WMMA helpers ----------------------------------------------------------

__device__ __forceinline__ v8f wmma_f16(v16h a, v16h b, v8f c) {
  return __builtin_amdgcn_wmma_f32_16x16x32_f16(
      /*neg_a=*/false, a, /*neg_b=*/false, b,
      /*c_mod=*/(short)0, c, /*reuse_a=*/false, /*reuse_b=*/false);
}

// A-fragment (16x32 f16) from an LDS tile [16 rows][SK halves], K-step kt.
// ISA layout: lane<16 -> row=lane, K = {kb..kb+7, 16+kb..16+kb+7}, kb=(lane>=16)*8
__device__ __forceinline__ v16h load_a(const _Float16* lds, int SK, int kt, int lane) {
  int r  = lane & 15;
  int kb = (lane >> 4) << 3;
  const _Float16* p = lds + r * SK + kt * 32 + kb;
  union { v16h v; v8h h[2]; } u;
  u.h[0] = *(const v8h*)(p);
  u.h[1] = *(const v8h*)(p + 16);
  return u.v;
}

// B-fragment: packed weights are laid out [kt][ct][lane][16 halves] contiguously.
__device__ __forceinline__ v16h load_b(const _Float16* pk) {
  return *(const v16h*)pk;
}

// ---- weight pre-pack: f32 [K][N] (or qkv [H][320][64]) -> f16 fragment layout
__global__ void pack_w(const float* __restrict__ src, _Float16* __restrict__ dst,
                       int K, int Ncols, int qkv_mode) {
  int p = blockIdx.x * blockDim.x + threadIdx.x;
  int total = K * Ncols;
  if (p >= total) return;
  int i    = p & 15;
  int lane = (p >> 4) & 31;
  int rest = p >> 9;                 // kt*NT + ct
  int NT   = Ncols >> 4;
  int ct   = rest % NT;
  int kt   = rest / NT;
  int kk   = kt * 32 + ((i >> 3) << 4) + ((lane >> 4) << 3) + (i & 7);   // K index
  int n    = ct * 16 + (lane & 15);              // col index
  float v;
  if (qkv_mode) {                    // src is [H][320][64], col = h*64+d
    int h = n >> 6;
    v = src[((size_t)(h * FIN + kk)) * 64 + (n & 63)];
  } else {
    v = src[(size_t)kk * Ncols + n];
  }
  dst[p] = (_Float16)v;
}

// ---------------------------------------------------------------------------
// Kernel 1: per-edge q/k/v GEMMs + softmax numerator + fused aggregation.
// 16 edges per block, 256 threads (8 waves), 48 output col-tiles (q|k|v).
// ---------------------------------------------------------------------------
__global__ void __launch_bounds__(256) edge_qkv_kernel(
    const float* __restrict__ node, const float* __restrict__ edge,
    const _Float16* __restrict__ WqP, const _Float16* __restrict__ WkP,
    const _Float16* __restrict__ WvP,
    const float* __restrict__ bq, const float* __restrict__ bk,
    const float* __restrict__ bv,
    const int* __restrict__ esrc, const int* __restrict__ edst,
    float* __restrict__ zsum, float* __restrict__ agg, int E) {
  __shared__ __align__(16) _Float16 fA[16 * SK_F];
  __shared__ __align__(16) float    qkv[16 * 768];   // q|k|v per edge row
  __shared__ float expd[16 * H];
  __shared__ int   sid[16], did[16];

  int tid = threadIdx.x;
  int e0  = blockIdx.x * 16;
  int ev  = min(16, E - e0);

  if (tid < 16) {
    int e = e0 + min(tid, ev - 1);
    sid[tid] = esrc[e];
    did[tid] = edst[e];
  }
  __syncthreads();

  // build f = [node[src] | edge | node[dst]]  (16 x 320, f16)
  for (int idx = tid; idx < 16 * FIN; idx += 256) {
    int r = idx / FIN, c = idx - r * FIN;
    int rr = min(r, ev - 1);
    float v;
    if (c < DN)            v = node[(size_t)sid[rr] * DN + c];
    else if (c < DN + DE)  v = edge[(size_t)(e0 + rr) * DE + (c - DN)];
    else                   v = node[(size_t)did[rr] * DN + (c - DN - DE)];
    fA[r * SK_F + c] = (_Float16)v;
  }
  __syncthreads();

  int wave = tid >> 5, lane = tid & 31;
  v8f acc[6];
#pragma unroll
  for (int t = 0; t < 6; ++t)
#pragma unroll
    for (int j = 0; j < 8; ++j) acc[t][j] = 0.f;

  const _Float16* wbase[6];
#pragma unroll
  for (int t = 0; t < 6; ++t) {
    int ct  = wave * 6 + t;                 // 0..47
    int mat = ct >> 4, ctl = ct & 15;
    const _Float16* W = (mat == 0) ? WqP : (mat == 1) ? WkP : WvP;
    wbase[t] = W + (size_t)ctl * 512 + lane * 16;
  }

  for (int kt = 0; kt < FIN / 32; ++kt) {   // 10 K-steps
    // batch: A + all 6 B fragments into independent regs, then WMMA burst
    v16h a = load_a(fA, SK_F, kt, lane);
    v16h b[6];
#pragma unroll
    for (int t = 0; t < 6; ++t)
      b[t] = load_b(wbase[t] + (size_t)kt * 8192);  // 16 tiles * 512 halves
#pragma unroll
    for (int t = 0; t < 6; ++t)
      acc[t] = wmma_f16(a, b[t], acc[t]);
  }

  // scatter C tiles into LDS (rows = edges, cols = 768)
#pragma unroll
  for (int t = 0; t < 6; ++t) {
    int ct  = wave * 6 + t;
    int col = ct * 16 + (lane & 15);
    int rb  = (lane >> 4) * 8;
#pragma unroll
    for (int j = 0; j < 8; ++j) qkv[(rb + j) * 768 + col] = acc[t][j];
  }
  __syncthreads();

  // dot/clip/exp per (edge, head), accumulate z
  if (tid < 16 * H) {
    int r = tid >> 2, h = tid & 3;
    float dot = 0.f;
#pragma unroll 8
    for (int d = 0; d < DK; ++d) {
      float qv = qkv[r * 768 +        h * DK + d] + bq[h * DK + d];
      float kv = qkv[r * 768 + 256  + h * DK + d] + bk[h * DK + d];
      dot += qv * kv;
    }
    dot *= RSQRT_FIN;
    dot  = fminf(5.f, fmaxf(-5.f, dot));
    float ex = __expf(dot);
    expd[r * H + h] = ex;
    if (r < ev) atomicAdd(&zsum[(size_t)did[r] * H + h], ex);
  }
  __syncthreads();

  // agg[dst] += exp * (vh + bv)   (unnormalized; divided by z in node kernel)
  for (int idx = tid; idx < 16 * DNO; idx += 256) {
    int r = idx >> 8, c = idx & 255;
    if (r >= ev) continue;
    int h = c >> 6;
    float vh = qkv[r * 768 + 512 + c] + bv[c];
    atomicAdd(&agg[(size_t)did[r] * DNO + c], expd[r * H + h] * vh);
  }
}

// ---------------------------------------------------------------------------
// Kernel 2: node pipeline. 16 nodes per block, 256 threads.
//   v = (agg/z)@Wvl + bvl ; res = node@Wres + bres ; x = LN(v+res)
//   h = LN(x@W1 + b1 + x) ; new_node = silu(h)@W2 + b2
// ---------------------------------------------------------------------------
__global__ void __launch_bounds__(256) node_kernel(
    const float* __restrict__ node,
    const float* __restrict__ agg, const float* __restrict__ zsum,
    const _Float16* __restrict__ WvlP, const _Float16* __restrict__ WresP,
    const _Float16* __restrict__ W1P,  const _Float16* __restrict__ W2P,
    const float* __restrict__ bvl, const float* __restrict__ bres,
    const float* __restrict__ b1,  const float* __restrict__ b2,
    const float* __restrict__ gA, const float* __restrict__ bA,
    const float* __restrict__ gF, const float* __restrict__ bF,
    float* __restrict__ out, int N) {
  __shared__ __align__(16) _Float16 xA[16 * SK_X];
  __shared__ __align__(16) float    tB[16 * DNO];
  __shared__ float red[512];
  __shared__ float mv[32];

  int tid = threadIdx.x, wave = tid >> 5, lane = tid & 31;
  int n0 = blockIdx.x * 16;

  // stage A = agg / z  (z==0 -> 1)
  for (int idx = tid; idx < 16 * DNO; idx += 256) {
    int r = idx >> 8, c = idx & 255;
    float z = zsum[(size_t)(n0 + r) * H + (c >> 6)];
    z = (z == 0.f) ? 1.f : z;
    xA[r * SK_X + c] = (_Float16)(agg[(size_t)(n0 + r) * DNO + c] / z);
  }
  __syncthreads();

  v8f acc[2];

  // GEMM1: v = A @ Wvl  (K=256; 16 col tiles -> 2 per wave)
#pragma unroll
  for (int t = 0; t < 2; ++t)
#pragma unroll
    for (int j = 0; j < 8; ++j) acc[t][j] = 0.f;
  for (int kt = 0; kt < 8; ++kt) {
    v16h a = load_a(xA, SK_X, kt, lane);
    v16h b[2];
#pragma unroll
    for (int t = 0; t < 2; ++t)
      b[t] = load_b(WvlP + ((size_t)(kt * 16 + wave * 2 + t) * 512) + lane * 16);
#pragma unroll
    for (int t = 0; t < 2; ++t) acc[t] = wmma_f16(a, b[t], acc[t]);
  }
#pragma unroll
  for (int t = 0; t < 2; ++t) {
    int col = (wave * 2 + t) * 16 + (lane & 15);
    int rb  = (lane >> 4) * 8;
#pragma unroll
    for (int j = 0; j < 8; ++j) tB[(rb + j) * DNO + col] = acc[t][j];
  }
  __syncthreads();

  // stage node features (16 x 128)
  for (int idx = tid; idx < 16 * DN; idx += 256) {
    int r = idx >> 7, c = idx & 127;
    xA[r * SK_X + c] = (_Float16)node[(size_t)(n0 + r) * DN + c];
  }
  __syncthreads();

  // GEMM2: residual = node @ Wres (K=128); epilogue adds biases into tB
#pragma unroll
  for (int t = 0; t < 2; ++t)
#pragma unroll
    for (int j = 0; j < 8; ++j) acc[t][j] = 0.f;
  for (int kt = 0; kt < 4; ++kt) {
    v16h a = load_a(xA, SK_X, kt, lane);
    v16h b[2];
#pragma unroll
    for (int t = 0; t < 2; ++t)
      b[t] = load_b(WresP + ((size_t)(kt * 16 + wave * 2 + t) * 512) + lane * 16);
#pragma unroll
    for (int t = 0; t < 2; ++t) acc[t] = wmma_f16(a, b[t], acc[t]);
  }
#pragma unroll
  for (int t = 0; t < 2; ++t) {
    int col = (wave * 2 + t) * 16 + (lane & 15);
    int rb  = (lane >> 4) * 8;
#pragma unroll
    for (int j = 0; j < 8; ++j)
      tB[(rb + j) * DNO + col] += acc[t][j] + bvl[col] + bres[col];
  }
  __syncthreads();

  // LN1 -> x (kept f32 in tB for the W1 skip, f16 in xA for the GEMM)
  {
    int r = tid >> 4, g = tid & 15;
    float s = 0.f, s2 = 0.f;
    for (int c = g * 16; c < g * 16 + 16; ++c) { float x = tB[r * DNO + c]; s += x; s2 += x * x; }
    red[r * 16 + g] = s; red[256 + r * 16 + g] = s2;
  }
  __syncthreads();
  if (tid < 16) {
    float S = 0.f, S2 = 0.f;
    for (int g = 0; g < 16; ++g) { S += red[tid * 16 + g]; S2 += red[256 + tid * 16 + g]; }
    float m = S / 256.f, var = S2 / 256.f - m * m;
    mv[tid * 2] = m; mv[tid * 2 + 1] = rsqrtf(var + EPS);
  }
  __syncthreads();
  for (int idx = tid; idx < 16 * DNO; idx += 256) {
    int r = idx >> 8, c = idx & 255;
    float x = (tB[r * DNO + c] - mv[r * 2]) * mv[r * 2 + 1] * gA[c] + bA[c];
    tB[r * DNO + c] = x;
    xA[r * SK_X + c] = (_Float16)x;
  }
  __syncthreads();

  // GEMM3: h_pre = x @ W1 + b1 + x
#pragma unroll
  for (int t = 0; t < 2; ++t)
#pragma unroll
    for (int j = 0; j < 8; ++j) acc[t][j] = 0.f;
  for (int kt = 0; kt < 8; ++kt) {
    v16h a = load_a(xA, SK_X, kt, lane);
    v16h b[2];
#pragma unroll
    for (int t = 0; t < 2; ++t)
      b[t] = load_b(W1P + ((size_t)(kt * 16 + wave * 2 + t) * 512) + lane * 16);
#pragma unroll
    for (int t = 0; t < 2; ++t) acc[t] = wmma_f16(a, b[t], acc[t]);
  }
#pragma unroll
  for (int t = 0; t < 2; ++t) {
    int col = (wave * 2 + t) * 16 + (lane & 15);
    int rb  = (lane >> 4) * 8;
#pragma unroll
    for (int j = 0; j < 8; ++j)
      tB[(rb + j) * DNO + col] += acc[t][j] + b1[col];
  }
  __syncthreads();

  // LN2 -> silu -> xA
  {
    int r = tid >> 4, g = tid & 15;
    float s = 0.f, s2 = 0.f;
    for (int c = g * 16; c < g * 16 + 16; ++c) { float x = tB[r * DNO + c]; s += x; s2 += x * x; }
    red[r * 16 + g] = s; red[256 + r * 16 + g] = s2;
  }
  __syncthreads();
  if (tid < 16) {
    float S = 0.f, S2 = 0.f;
    for (int g = 0; g < 16; ++g) { S += red[tid * 16 + g]; S2 += red[256 + tid * 16 + g]; }
    float m = S / 256.f, var = S2 / 256.f - m * m;
    mv[tid * 2] = m; mv[tid * 2 + 1] = rsqrtf(var + EPS);
  }
  __syncthreads();
  for (int idx = tid; idx < 16 * DNO; idx += 256) {
    int r = idx >> 8, c = idx & 255;
    float x = (tB[r * DNO + c] - mv[r * 2]) * mv[r * 2 + 1] * gF[c] + bF[c];
    float sl = x / (1.f + __expf(-x));
    xA[r * SK_X + c] = (_Float16)sl;
  }
  __syncthreads();

  // GEMM4: new_node = silu(h) @ W2 + b2 -> d_out
#pragma unroll
  for (int t = 0; t < 2; ++t)
#pragma unroll
    for (int j = 0; j < 8; ++j) acc[t][j] = 0.f;
  for (int kt = 0; kt < 8; ++kt) {
    v16h a = load_a(xA, SK_X, kt, lane);
    v16h b[2];
#pragma unroll
    for (int t = 0; t < 2; ++t)
      b[t] = load_b(W2P + ((size_t)(kt * 16 + wave * 2 + t) * 512) + lane * 16);
#pragma unroll
    for (int t = 0; t < 2; ++t) acc[t] = wmma_f16(a, b[t], acc[t]);
  }
#pragma unroll
  for (int t = 0; t < 2; ++t) {
    int col = (wave * 2 + t) * 16 + (lane & 15);
    int rb  = (lane >> 4) * 8;
#pragma unroll
    for (int j = 0; j < 8; ++j)
      out[(size_t)(n0 + rb + j) * DNO + col] = acc[t][j] + b2[col];
  }
}

// ---------------------------------------------------------------------------
// Kernel 3: edge output. 16 edges per block.
//   ef = [nn[src] | edge | nn[dst]] (576) ; e = LN(ef@W3+b3) ; out = silu(e)@W4+b4
// ---------------------------------------------------------------------------
__global__ void __launch_bounds__(256) edge_out_kernel(
    const float* __restrict__ nn, const float* __restrict__ edge,
    const _Float16* __restrict__ W3P, const _Float16* __restrict__ W4P,
    const float* __restrict__ b3, const float* __restrict__ b4,
    const float* __restrict__ gE, const float* __restrict__ bE,
    const int* __restrict__ esrc, const int* __restrict__ edst,
    float* __restrict__ oute, int E) {
  __shared__ __align__(16) _Float16 eA[16 * SK_EF];
  __shared__ __align__(16) float    eB[16 * DOE];
  __shared__ float red[512];
  __shared__ float mv[32];
  __shared__ int   sid[16], did[16];

  int tid = threadIdx.x, wave = tid >> 5, lane = tid & 31;
  int e0 = blockIdx.x * 16;
  int ev = min(16, E - e0);

  if (tid < 16) {
    int e = e0 + min(tid, ev - 1);
    sid[tid] = esrc[e];
    did[tid] = edst[e];
  }
  __syncthreads();

  for (int idx = tid; idx < 16 * FIN2; idx += 256) {
    int r = idx / FIN2, c = idx - r * FIN2;
    int rr = min(r, ev - 1);
    float v;
    if (c < DNO)            v = nn[(size_t)sid[rr] * DNO + c];
    else if (c < DNO + DE)  v = edge[(size_t)(e0 + rr) * DE + (c - DNO)];
    else                    v = nn[(size_t)did[rr] * DNO + (c - DNO - DE)];
    eA[r * SK_EF + c] = (_Float16)v;
  }
  __syncthreads();

  // GEMM: e_pre = ef @ W3 (K=576, 8 col tiles -> 1 per wave, 18 K-steps)
  // 2-way K unroll keeps two A/B pairs in flight per iteration.
  v8f acc;
#pragma unroll
  for (int j = 0; j < 8; ++j) acc[j] = 0.f;
  for (int kt = 0; kt < FIN2 / 32; kt += 2) {
    v16h a0 = load_a(eA, SK_EF, kt,     lane);
    v16h a1 = load_a(eA, SK_EF, kt + 1, lane);
    v16h b0 = load_b(W3P + ((size_t)((kt    ) * 8 + wave) * 512) + lane * 16);
    v16h b1 = load_b(W3P + ((size_t)((kt + 1) * 8 + wave) * 512) + lane * 16);
    acc = wmma_f16(a0, b0, acc);
    acc = wmma_f16(a1, b1, acc);
  }
  {
    int col = wave * 16 + (lane & 15);
    int rb  = (lane >> 4) * 8;
#pragma unroll
    for (int j = 0; j < 8; ++j) eB[(rb + j) * DOE + col] = acc[j] + b3[col];
  }
  __syncthreads();

  // LN over 128 (gE,bE)
  {
    int r = tid >> 4, g = tid & 15;
    float s = 0.f, s2 = 0.f;
    for (int c = g * 8; c < g * 8 + 8; ++c) { float x = eB[r * DOE + c]; s += x; s2 += x * x; }
    red[r * 16 + g] = s; red[256 + r * 16 + g] = s2;
  }
  __syncthreads();
  if (tid < 16) {
    float S = 0.f, S2 = 0.f;
    for (int g = 0; g < 16; ++g) { S += red[tid * 16 + g]; S2 += red[256 + tid * 16 + g]; }
    float m = S / 128.f, var = S2 / 128.f - m * m;
    mv[tid * 2] = m; mv[tid * 2 + 1] = rsqrtf(var + EPS);
  }
  __syncthreads();
  for (int idx = tid; idx < 16 * DOE; idx += 256) {
    int r = idx >> 7, c = idx & 127;
    float x = (eB[r * DOE + c] - mv[r * 2]) * mv[r * 2 + 1] * gE[c] + bE[c];
    float sl = x / (1.f + __expf(-x));
    eA[r * SK_EF + c] = (_Float16)sl;
  }
  __syncthreads();

  // GEMM: new_edge = silu(e) @ W4 + b4 (K=128, 4 K-steps, 2-way unroll)
#pragma unroll
  for (int j = 0; j < 8; ++j) acc[j] = 0.f;
  for (int kt = 0; kt < 4; kt += 2) {
    v16h a0 = load_a(eA, SK_EF, kt,     lane);
    v16h a1 = load_a(eA, SK_EF, kt + 1, lane);
    v16h b0 = load_b(W4P + ((size_t)((kt    ) * 8 + wave) * 512) + lane * 16);
    v16h b1 = load_b(W4P + ((size_t)((kt + 1) * 8 + wave) * 512) + lane * 16);
    acc = wmma_f16(a0, b0, acc);
    acc = wmma_f16(a1, b1, acc);
  }
  {
    int col = wave * 16 + (lane & 15);
    int rb  = (lane >> 4) * 8;
#pragma unroll
    for (int j = 0; j < 8; ++j) {
      int r = rb + j;
      if (r < ev) oute[(size_t)(e0 + r) * DOE + col] = acc[j] + b4[col];
    }
  }
}

// ---------------------------------------------------------------------------
// Host launcher
// ---------------------------------------------------------------------------
extern "C" void kernel_launch(void* const* d_in, const int* in_sizes, int n_in,
                              void* d_out, int out_size, void* d_ws, size_t ws_size,
                              hipStream_t stream) {
  (void)n_in; (void)out_size; (void)ws_size;
  const float* node = (const float*)d_in[0];
  const float* edge = (const float*)d_in[1];
  const float* Wq   = (const float*)d_in[2];  const float* bq   = (const float*)d_in[3];
  const float* Wk   = (const float*)d_in[4];  const float* bk   = (const float*)d_in[5];
  const float* Wv   = (const float*)d_in[6];  const float* bv   = (const float*)d_in[7];
  const float* Wvl  = (const float*)d_in[8];  const float* bvl  = (const float*)d_in[9];
  const float* Wres = (const float*)d_in[10]; const float* bres = (const float*)d_in[11];
  const float* W1   = (const float*)d_in[12]; const float* b1   = (const float*)d_in[13];
  const float* W2   = (const float*)d_in[14]; const float* b2   = (const float*)d_in[15];
  const float* W3   = (const float*)d_in[16]; const float* b3   = (const float*)d_in[17];
  const float* W4   = (const float*)d_in[18]; const float* b4   = (const float*)d_in[19];
  const float* gA   = (const float*)d_in[20]; const float* bA   = (const float*)d_in[21];
  const float* gF   = (const float*)d_in[22]; const float* bF   = (const float*)d_in[23];
  const float* gE   = (const float*)d_in[24]; const float* bE   = (const float*)d_in[25];
  const int*   esrc = (const int*)d_in[26];
  const int*   edst = (const int*)d_in[27];

  int N = in_sizes[0] / DN;
  int E = in_sizes[1] / DE;

  float* out_node = (float*)d_out;
  float* out_edge = out_node + (size_t)N * DNO;

  // workspace carve-out (256B aligned regions)
  char*  ws  = (char*)d_ws;
  size_t off = 0;
  auto carve = [&](size_t bytes) { size_t o = off; off = (off + bytes + 255) & ~(size_t)255; return o; };

  size_t zoff   = carve((size_t)N * H * 4);        // zsum f32 [N,H]
  size_t aoff   = carve((size_t)N * DNO * 4);      // agg  f32 [N,256]
  size_t zeroEnd = off;                            // [0, zeroEnd) zeroed each call
  size_t wqoff  = carve((size_t)FIN * DNO * 2);    // packed f16 weights
  size_t wkoff  = carve((size_t)FIN * DNO * 2);
  size_t wvoff  = carve((size_t)FIN * DNO * 2);
  size_t wvloff = carve((size_t)DNO * DNO * 2);
  size_t wroff  = carve((size_t)DN  * DNO * 2);
  size_t w1off  = carve((size_t)DNO * DNO * 2);
  size_t w2off  = carve((size_t)DNO * DNO * 2);
  size_t w3off  = carve((size_t)FIN2 * DOE * 2);
  size_t w4off  = carve((size_t)DOE * DOE * 2);

  float*    zsum = (float*)(ws + zoff);
  float*    agg  = (float*)(ws + aoff);
  _Float16* WqP  = (_Float16*)(ws + wqoff);
  _Float16* WkP  = (_Float16*)(ws + wkoff);
  _Float16* WvP  = (_Float16*)(ws + wvoff);
  _Float16* WvlP = (_Float16*)(ws + wvloff);
  _Float16* WresP= (_Float16*)(ws + wroff);
  _Float16* W1P  = (_Float16*)(ws + w1off);
  _Float16* W2P  = (_Float16*)(ws + w2off);
  _Float16* W3P  = (_Float16*)(ws + w3off);
  _Float16* W4P  = (_Float16*)(ws + w4off);

  hipMemsetAsync(ws, 0, zeroEnd, stream);

  auto packLaunch = [&](const float* src, _Float16* dst, int K, int Ncols, int mode) {
    int total = K * Ncols;
    pack_w<<<(total + 255) / 256, 256, 0, stream>>>(src, dst, K, Ncols, mode);
  };
  packLaunch(Wq,   WqP,   FIN,  DNO, 1);
  packLaunch(Wk,   WkP,   FIN,  DNO, 1);
  packLaunch(Wv,   WvP,   FIN,  DNO, 1);
  packLaunch(Wvl,  WvlP,  DNO,  DNO, 0);
  packLaunch(Wres, WresP, DN,   DNO, 0);
  packLaunch(W1,   W1P,   DNO,  DNO, 0);
  packLaunch(W2,   W2P,   DNO,  DNO, 0);
  packLaunch(W3,   W3P,   FIN2, DOE, 0);
  packLaunch(W4,   W4P,   DOE,  DOE, 0);

  int eblocks = (E + 15) / 16;
  int nblocks = (N + 15) / 16;

  edge_qkv_kernel<<<eblocks, 256, 0, stream>>>(
      node, edge, WqP, WkP, WvP, bq, bk, bv, esrc, edst, zsum, agg, E);

  node_kernel<<<nblocks, 256, 0, stream>>>(
      node, agg, zsum, WvlP, WresP, W1P, W2P,
      bvl, bres, b1, b2, gA, bA, gF, bF, out_node, N);

  edge_out_kernel<<<eblocks, 256, 0, stream>>>(
      out_node, edge, W3P, W4P, b3, b4, gE, bE, esrc, edst, out_edge, E);
}
